// GroupLevelGNN_70342974374329
// MI455X (gfx1250) — matrix operand order repaired and checked by reading, not compiled, not run on hardware
//
#include <hip/hip_runtime.h>

// ---------------------------------------------------------------------------
// GroupLevelGNN for MI455X (gfx1250): bf16 WMMA path, rotated 2-deep pipeline
// ---------------------------------------------------------------------------

#define G_GROUPS 4096
#define K_ATOMS  16
#define N_ATOMS  16384
#define A_DIM    256
#define F_DIM    128
#define H_DIM    256
#define L_LAYERS 2
#define XK       (A_DIM + F_DIM)   // 384 concatenated input K

typedef __attribute__((ext_vector_type(16))) __bf16 v16bf;
typedef __attribute__((ext_vector_type(8)))  float  v8f;

union Frag16 {
    v16bf v;
    uint4 q[2];
};

__device__ __forceinline__ unsigned short f2bf(float x) {
    union { float f; unsigned u; } v; v.f = x;
    unsigned r = v.u + 0x7FFFu + ((v.u >> 16) & 1u);   // round-to-nearest-even
    return (unsigned short)(r >> 16);
}

// ---------------------------------------------------------------------------
// Workspace layout (bytes)
// ---------------------------------------------------------------------------
static constexpr size_t OFF_ADJ   = 0;                              // 4096*4096*2
static constexpr size_t OFF_X     = OFF_ADJ   + 33554432;           // 4096*384*2
static constexpr size_t OFF_WCAT  = OFF_X     + 3145728;            // 256*384*2
static constexpr size_t OFF_WL    = OFF_WCAT  + 196608;             // 4*256*256*2
static constexpr size_t OFF_BIAS0 = OFF_WL    + 524288;             // 256*4
static constexpr size_t OFF_GEA   = OFF_BIAS0 + 1024;               // 4096*256*2
static constexpr size_t OFF_GETA  = OFF_GEA   + 2097152;
static constexpr size_t OFF_GEB   = OFF_GETA  + 2097152;
static constexpr size_t OFF_GETB  = OFF_GEB   + 2097152;
static constexpr size_t OFF_MSG   = OFF_GETB  + 2097152;
static constexpr size_t OFF_BLOOM = OFF_MSG   + 2097152;            // 4096*8

// ---------------------------------------------------------------------------
// Prep: weights -> bf16 BT ([N][K]) layouts, fused input bias
// ---------------------------------------------------------------------------
__global__ void prep_weights_kernel(const float* __restrict__ W_in,
                                    const float* __restrict__ b_in,
                                    const float* __restrict__ W_a2g,
                                    const float* __restrict__ b_a2g,
                                    const float* __restrict__ W_self,
                                    const float* __restrict__ W_neigh,
                                    unsigned short* __restrict__ WcatT,
                                    unsigned short* __restrict__ WlT,
                                    float* __restrict__ bias0) {
    int id = blockIdx.x * 256 + threadIdx.x;
    const int nWcat = H_DIM * XK;           // 98304
    const int nWl   = 4 * H_DIM * H_DIM;    // 262144
    if (id < nWcat) {
        int n = id / XK, k = id % XK;
        float w = (k < A_DIM) ? W_a2g[k * H_DIM + n] : W_in[(k - A_DIM) * H_DIM + n];
        WcatT[(size_t)n * XK + k] = f2bf(w);
        return;
    }
    id -= nWcat;
    if (id < nWl) {
        int mat = id / (H_DIM * H_DIM);
        int e   = id % (H_DIM * H_DIM);
        int n = e / H_DIM, k = e % H_DIM;
        const float* src = ((mat & 1) ? W_neigh : W_self) + (size_t)(mat >> 1) * H_DIM * H_DIM;
        WlT[(size_t)mat * H_DIM * H_DIM + (size_t)n * H_DIM + k] = f2bf(src[(size_t)k * H_DIM + n]);
        return;
    }
    id -= nWl;
    if (id < H_DIM) bias0[id] = b_a2g[id] + b_in[id];
}

// ---------------------------------------------------------------------------
// Bloom: 64-bit membership mask per group
// ---------------------------------------------------------------------------
__global__ void bloom_kernel(const int* __restrict__ gidx,
                             unsigned long long* __restrict__ bloom) {
    int g = blockIdx.x * 256 + threadIdx.x;
    if (g >= G_GROUPS) return;
    unsigned long long b = 0;
    #pragma unroll
    for (int j = 0; j < K_ATOMS; ++j)
        b |= 1ull << (gidx[g * K_ATOMS + j] & 63);
    bloom[g] = b;
}

// ---------------------------------------------------------------------------
// Pool: gather 16 atom rows, mean, concat group features -> bf16 [G, 384]
// ---------------------------------------------------------------------------
__global__ void pool_kernel(const float* __restrict__ atom,
                            const int* __restrict__ gidx,
                            const float* __restrict__ gf,
                            unsigned short* __restrict__ Xbf) {
    int g = blockIdx.x;
    int t = threadIdx.x;
    __shared__ int ids[K_ATOMS];
    if (t < K_ATOMS) ids[t] = gidx[g * K_ATOMS + t];
    __syncthreads();
    float s = 0.f;
    #pragma unroll
    for (int j = 0; j < K_ATOMS; ++j)
        s += atom[(size_t)ids[j] * A_DIM + t];
    Xbf[(size_t)g * XK + t] = f2bf(s * (1.0f / K_ATOMS));
    if (t < F_DIM)
        Xbf[(size_t)g * XK + A_DIM + t] = f2bf(gf[(size_t)g * F_DIM + t]);
}

// ---------------------------------------------------------------------------
// Adjacency: bloom-prefiltered exact 16x16 set intersection -> bf16 {0,1}
// ---------------------------------------------------------------------------
__global__ void adj_kernel(const int* __restrict__ gidx,
                           const unsigned long long* __restrict__ bloom,
                           unsigned short* __restrict__ adj) {
    int g = blockIdx.x;
    __shared__ int my[K_ATOMS];
    __shared__ unsigned long long bgs;
    if (threadIdx.x < K_ATOMS) my[threadIdx.x] = gidx[g * K_ATOMS + threadIdx.x];
    if (threadIdx.x == 0) bgs = bloom[g];
    __syncthreads();
    unsigned long long bg = bgs;
    int h0 = threadIdx.x * 16;
    union { unsigned short s[16]; uint4 q[2]; } row;
    #pragma unroll
    for (int j = 0; j < 16; ++j) {
        int h = h0 + j;
        unsigned short v = 0;
        if (h != g && (bloom[h] & bg) != 0ull) {
            const int* hid = gidx + (size_t)h * K_ATOMS;
            bool hit = false;
            #pragma unroll
            for (int a = 0; a < K_ATOMS; ++a) {
                int idv = hid[a];
                #pragma unroll
                for (int b = 0; b < K_ATOMS; ++b) hit |= (idv == my[b]);
            }
            v = hit ? (unsigned short)0x3F80 : (unsigned short)0;   // bf16 1.0
        }
        row.s[j] = v;
    }
    uint4* dst = reinterpret_cast<uint4*>(adj + (size_t)g * G_GROUPS + h0);
    dst[0] = row.q[0];
    dst[1] = row.q[1];
}

// ---------------------------------------------------------------------------
// Generic bf16 WMMA GEMM: D = A1*B1 (+ A2*B2) (+bias) (ReLU)
//   A  : row-major [M][K] bf16,  BT : row-major [N][K] bf16 (B column-major)
//   REQUIRES K % 64 == 0 (all call sites: 384 / 4096 / 256).
// One wave computes a 16x64 tile with 4 accumulators reusing the A fragment.
// Two fragment register sets rotate branch-free: while set0's 4 WMMAs issue,
// set1's loads for k+32 are in flight, and vice versa — no acc copies, no
// full loadcnt drains.
// ---------------------------------------------------------------------------
__device__ __forceinline__ void load_frags(const unsigned short* __restrict__ arow,
                                           const unsigned short* __restrict__ BT,
                                           int ldb, int colbase, int k0,
                                           int mn, int kh,
                                           Frag16& a, Frag16 b[4]) {
    a.q[0] = *reinterpret_cast<const uint4*>(arow + k0 + kh * 8);        // VGPR 0-3
    a.q[1] = *reinterpret_cast<const uint4*>(arow + k0 + 16 + kh * 8);   // VGPR 4-7
    #pragma unroll
    for (int q = 0; q < 4; ++q) {
        const unsigned short* bp =
            BT + (size_t)(colbase + q * 16 + mn) * ldb + k0 + kh * 8;
        b[q].q[0] = *reinterpret_cast<const uint4*>(bp);
        b[q].q[1] = *reinterpret_cast<const uint4*>(bp + 16);
    }
}

__device__ __forceinline__ void wmma4(const Frag16& a, const Frag16 b[4], v8f acc[4]) {
    #pragma unroll
    for (int q = 0; q < 4; ++q)
        acc[q] = __builtin_amdgcn_wmma_f32_16x16x32_bf16(
            false, a.v, false, b[q].v, (short)0, acc[q], false, false);
}

__device__ __forceinline__ void gemm_seg(const unsigned short* __restrict__ A, int lda,
                                         const unsigned short* __restrict__ BT, int ldb,
                                         int K, int tm, int tnq, int lane, v8f acc[4]) {
    const int mn = lane & 15;   // A row within tile == B col within tile
    const int kh = lane >> 4;   // K half select
    const int colbase = tnq * 64;
    const unsigned short* arow = A + (size_t)(tm * 16 + mn) * lda;

    Frag16 a0, b0[4], a1, b1[4];
    load_frags(arow, BT, ldb, colbase, 0,  mn, kh, a0, b0);
    load_frags(arow, BT, ldb, colbase, 32, mn, kh, a1, b1);

    int k0 = 0;
    for (; k0 + 64 < K; k0 += 64) {
        wmma4(a0, b0, acc);
        load_frags(arow, BT, ldb, colbase, k0 + 64, mn, kh, a0, b0);
        wmma4(a1, b1, acc);
        load_frags(arow, BT, ldb, colbase, k0 + 96, mn, kh, a1, b1);
    }
    wmma4(a0, b0, acc);   // k0
    wmma4(a1, b1, acc);   // k0 + 32
}

__global__ void __launch_bounds__(256) wmma_gemm_kernel(
    const unsigned short* __restrict__ A1, int lda1, int K1,
    const unsigned short* __restrict__ BT1, int ldb1,
    const unsigned short* __restrict__ A2, int lda2, int K2,
    const unsigned short* __restrict__ BT2, int ldb2,
    const float* __restrict__ bias, int relu,
    float* __restrict__ outF,
    unsigned short* __restrict__ outB,
    unsigned short* __restrict__ outT,
    int M, int N) {
    const int lane = threadIdx.x & 31;
    const int wave = threadIdx.x >> 5;
    const int Ntq = N >> 6;                 // 64-wide column groups
    const int T = (M >> 4) * Ntq;
    int t = blockIdx.x * 8 + wave;
    if (t >= T) return;
    const int tm = t / Ntq;
    const int tnq = t - tm * Ntq;

    v8f acc[4] = {};
    gemm_seg(A1, lda1, BT1, ldb1, K1, tm, tnq, lane, acc);
    if (A2) gemm_seg(A2, lda2, BT2, ldb2, K2, tm, tnq, lane, acc);

    const int n = lane & 15;
    const int kh = lane >> 4;
    const int rowbase = tm * 16 + kh * 8;   // C/D layout: VGPR r -> row kh*8+r
    #pragma unroll
    for (int q = 0; q < 4; ++q) {
        const int col = tnq * 64 + q * 16 + n;
        const float bv = bias ? bias[col] : 0.0f;
        union { unsigned short s[8]; uint4 q4[2]; } pk;
        #pragma unroll
        for (int r = 0; r < 8; ++r) {
            float x = acc[q][r] + bv;
            if (relu) x = fmaxf(x, 0.0f);
            const size_t idx = (size_t)(rowbase + r) * N + col;
            if (outF) outF[idx] = x;
            if (outB) outB[idx] = f2bf(x);
            pk.s[r] = f2bf(x);
        }
        if (outT) {   // transposed bf16: 8 consecutive rows of column `col`
            *reinterpret_cast<uint4*>(outT + (size_t)col * M + rowbase) = pk.q4[0];
        }
    }
}

// ---------------------------------------------------------------------------
// Host: launch pipeline
// ---------------------------------------------------------------------------
extern "C" void kernel_launch(void* const* d_in, const int* in_sizes, int n_in,
                              void* d_out, int out_size, void* d_ws, size_t ws_size,
                              hipStream_t stream) {
    (void)in_sizes; (void)n_in; (void)out_size; (void)ws_size;

    const float* atom    = (const float*)d_in[0];
    const int*   gidx    = (const int*)  d_in[1];
    const float* gf      = (const float*)d_in[2];
    const float* W_in    = (const float*)d_in[3];
    const float* b_in    = (const float*)d_in[4];
    const float* W_a2g   = (const float*)d_in[5];
    const float* b_a2g   = (const float*)d_in[6];
    const float* W_self  = (const float*)d_in[7];
    const float* W_neigh = (const float*)d_in[8];
    const float* b_mp    = (const float*)d_in[9];
    float* out = (float*)d_out;

    char* ws = (char*)d_ws;
    unsigned short* adjbf = (unsigned short*)(ws + OFF_ADJ);
    unsigned short* Xbf   = (unsigned short*)(ws + OFF_X);
    unsigned short* WcatT = (unsigned short*)(ws + OFF_WCAT);
    unsigned short* WlT   = (unsigned short*)(ws + OFF_WL);
    unsigned short* WsT0  = WlT;
    unsigned short* WnT0  = WlT + 1 * H_DIM * H_DIM;
    unsigned short* WsT1  = WlT + 2 * H_DIM * H_DIM;
    unsigned short* WnT1  = WlT + 3 * H_DIM * H_DIM;
    float*          bias0 = (float*)(ws + OFF_BIAS0);
    unsigned short* gebfA = (unsigned short*)(ws + OFF_GEA);
    unsigned short* geTA  = (unsigned short*)(ws + OFF_GETA);
    unsigned short* gebfB = (unsigned short*)(ws + OFF_GEB);
    unsigned short* geTB  = (unsigned short*)(ws + OFF_GETB);
    unsigned short* msgbf = (unsigned short*)(ws + OFF_MSG);
    unsigned long long* bloom = (unsigned long long*)(ws + OFF_BLOOM);

    // ---- prep ----
    prep_weights_kernel<<<1409, 256, 0, stream>>>(W_in, b_in, W_a2g, b_a2g,
                                                  W_self, W_neigh, WcatT, WlT, bias0);
    bloom_kernel<<<16, 256, 0, stream>>>(gidx, bloom);
    pool_kernel<<<G_GROUPS, 256, 0, stream>>>(atom, gidx, gf, Xbf);
    adj_kernel<<<G_GROUPS, 256, 0, stream>>>(gidx, bloom, adjbf);

    const int gemmBlocks = ((G_GROUPS / 16) * (H_DIM / 64) + 7) / 8;   // 128

    // ---- ge0 = [pooled | gf] @ [W_a2g; W_in] + (b_a2g + b_in) ----
    wmma_gemm_kernel<<<gemmBlocks, 256, 0, stream>>>(
        Xbf, XK, XK, WcatT, XK,
        nullptr, 0, 0, nullptr, 0,
        bias0, 0,
        nullptr, gebfA, geTA, G_GROUPS, H_DIM);

    // ---- layer 0 ----
    wmma_gemm_kernel<<<gemmBlocks, 256, 0, stream>>>(
        adjbf, G_GROUPS, G_GROUPS, geTA, G_GROUPS,
        nullptr, 0, 0, nullptr, 0,
        nullptr, 0,
        nullptr, msgbf, nullptr, G_GROUPS, H_DIM);
    wmma_gemm_kernel<<<gemmBlocks, 256, 0, stream>>>(
        gebfA, H_DIM, H_DIM, WsT0, H_DIM,
        msgbf, H_DIM, H_DIM, WnT0, H_DIM,
        b_mp, 1,
        nullptr, gebfB, geTB, G_GROUPS, H_DIM);

    // ---- layer 1 ----
    wmma_gemm_kernel<<<gemmBlocks, 256, 0, stream>>>(
        adjbf, G_GROUPS, G_GROUPS, geTB, G_GROUPS,
        nullptr, 0, 0, nullptr, 0,
        nullptr, 0,
        nullptr, msgbf, nullptr, G_GROUPS, H_DIM);
    wmma_gemm_kernel<<<gemmBlocks, 256, 0, stream>>>(
        gebfB, H_DIM, H_DIM, WsT1, H_DIM,
        msgbf, H_DIM, H_DIM, WnT1, H_DIM,
        b_mp + H_DIM, 1,
        out, nullptr, nullptr, G_GROUPS, H_DIM);
}
